// SpatialAttention_65154653880379
// MI455X (gfx1250) — matrix operand order repaired and checked by reading, not compiled, not run on hardware
//
#include <hip/hip_runtime.h>
#include <hip/hip_bf16.h>
#include <math.h>

typedef __attribute__((ext_vector_type(16))) __bf16 v16bf;
typedef __attribute__((ext_vector_type(8)))  __bf16 v8bf;
typedef __attribute__((ext_vector_type(8)))  float  v8f;

#define B_   2
#define S_   2048
#define T_   8
#define C_   128
#define K_   16
#define H_   4
#define D_   32
#define BT_  (B_ * T_)
#define CP_  160                 // C+2=130 padded up to a multiple of 32
#define STILE 16
#define NSTILE (S_ / STILE)      // 128

// bf16 weight workspace layout (element offsets)
#define WQ_OFF 0
#define WK_OFF (C_ * C_)             // 16384
#define WV_OFF (WK_OFF + C_ * CP_)   // 36864
#define WP_OFF (WV_OFF + C_ * CP_)   // 57344
#define WTOT   (WP_OFF + C_ * C_)    // 73728 bf16 elems = 147456 bytes

// ---------------------------------------------------------------------------
// Prep: convert fp32 weights to bf16, zero-pad Wk/Wv feature dim 130 -> 160
// ---------------------------------------------------------------------------
__global__ void prep_weights_kernel(const float* __restrict__ Wq,
                                    const float* __restrict__ Wk,
                                    const float* __restrict__ Wv,
                                    const float* __restrict__ Wp,
                                    __bf16* __restrict__ wbf) {
    int e = blockIdx.x * blockDim.x + threadIdx.x;
    if (e >= WTOT) return;
    float v;
    if (e < WK_OFF) {
        v = Wq[e];
    } else if (e < WV_OFF) {
        int i = e - WK_OFF; int r = i / CP_, c = i % CP_;
        v = (c < C_ + 2) ? Wk[r * (C_ + 2) + c] : 0.0f;
    } else if (e < WP_OFF) {
        int i = e - WV_OFF; int r = i / CP_, c = i % CP_;
        v = (c < C_ + 2) ? Wv[r * (C_ + 2) + c] : 0.0f;
    } else {
        v = Wp[e - WP_OFF];
    }
    wbf[e] = (__bf16)v;
}

// ---------------------------------------------------------------------------
// Fused spatial attention: one block = one (bt, 16-row s tile), 4 waves
// ---------------------------------------------------------------------------
__global__ __launch_bounds__(128)
void spatial_attn_kernel(const float* __restrict__ x,
                         const int*   __restrict__ sidx,
                         const float* __restrict__ swgt,
                         const float* __restrict__ ali,
                         const float* __restrict__ dst,
                         const __bf16* __restrict__ wbf,
                         const float* __restrict__ bp,
                         float* __restrict__ out) {
    __shared__ __bf16 xtile[STILE][C_];        //  4 KB  x rows, bf16
    __shared__ float  qtile[STILE][C_];        //  8 KB  Q tile, fp32
    __shared__ __bf16 knLds[4][K_][C_];        // 16 KB  per-wave kn
    __shared__ __bf16 vnLds[4][K_][C_];        // 16 KB  per-wave vn
    __shared__ float  attnLds[4][H_][K_];      //  1 KB  per-wave softmax
    __shared__ float  otile[STILE][C_];        //  8 KB  attention output

    const int tid  = threadIdx.x;
    const int lane = tid & 31;
    const int w    = tid >> 5;                 // wave 0..3
    const int bt    = blockIdx.x / NSTILE;
    const int stile = blockIdx.x % NSTILE;
    const int s0    = stile * STILE;
    const int b = bt / T_;
    const int t = bt % T_;

    const __bf16* wq = wbf + WQ_OFF;
    const __bf16* wk = wbf + WK_OFF;
    const __bf16* wv = wbf + WV_OFF;
    const __bf16* wp = wbf + WP_OFF;

    const int mrow = lane & 15;                // A-matrix row / neighbor id
    const int base = (lane >> 4) * 8;          // C/D-fragment row base (8 * lane half)
    const int ncol = lane & 15;                // B/C/D column within tile
    const int kseg = (lane >> 4) * 16;         // B-fragment K run base

    // ---- Stage 1: stage x tile (16 s rows x 128 ch) into LDS as bf16 ----
    {
        int row = tid >> 3;                    // 0..15
        int c16 = (tid & 7) * 16;              // 0..112
        const float* src = x + ((size_t)(b * S_ + s0 + row) * T_ + t) * C_ + c16;
        #pragma unroll
        for (int q4 = 0; q4 < 4; ++q4) {
            float4 f = *(const float4*)(src + q4 * 4);
            __bf16* d = &xtile[row][c16 + q4 * 4];
            d[0] = (__bf16)f.x; d[1] = (__bf16)f.y;
            d[2] = (__bf16)f.z; d[3] = (__bf16)f.w;
        }
    }
    __syncthreads();

    // ---- Stage 2: Q = Xtile * Wq^T via WMMA (wave w owns n-tiles 2w,2w+1) --
    {
        v16bf axf[4];
        #pragma unroll
        for (int ks = 0; ks < 4; ++ks) {
            v8bf a0 = *(const v8bf*)&xtile[mrow][ks * 32 + kseg];
            v8bf a1 = *(const v8bf*)&xtile[mrow][ks * 32 + kseg + 8];
            #pragma unroll
            for (int j = 0; j < 8; ++j) { axf[ks][j] = a0[j]; axf[ks][8 + j] = a1[j]; }
        }
        #pragma unroll
        for (int nt = 0; nt < 2; ++nt) {
            const int n = w * 2 + nt;
            v8f acc = {0.f, 0.f, 0.f, 0.f, 0.f, 0.f, 0.f, 0.f};
            #pragma unroll
            for (int ks = 0; ks < 4; ++ks) {
                v16bf bfr = *(const v16bf*)(wq + (size_t)(n * 16 + ncol) * C_ + ks * 32 + kseg);
                acc = __builtin_amdgcn_wmma_f32_16x16x32_bf16(
                        false, axf[ks], false, bfr, (short)0, acc, false, false);
            }
            #pragma unroll
            for (int j = 0; j < 8; ++j)
                qtile[base + j][n * 16 + ncol] = acc[j];
        }
    }
    __syncthreads();

    const float rscale = 0.17677669529663687f;   // 1/sqrt(D=32)

    // ---- Stage 3/4: per-s neighbor projections + attention (4 s per wave) --
    for (int si = 0; si < 4; ++si) {
        const int srow = w * 4 + si;
        const int s    = s0 + srow;

        // Gather neighbor row mrow and pack 5 bf16 A-fragments (K padded 160)
        const int    nb      = sidx[((size_t)bt * S_ + s) * K_ + mrow];
        const size_t rowBase = ((size_t)(b * S_ + nb) * T_ + t) * C_;
        v16bf afr[5];
        #pragma unroll
        for (int ks = 0; ks < 4; ++ks) {
            const float* p0 = x + rowBase + ks * 32 + kseg;
            float4 f00 = *(const float4*)(p0);
            float4 f01 = *(const float4*)(p0 + 4);
            float4 f10 = *(const float4*)(p0 + 8);
            float4 f11 = *(const float4*)(p0 + 12);
            afr[ks][0]  = (__bf16)f00.x; afr[ks][1]  = (__bf16)f00.y;
            afr[ks][2]  = (__bf16)f00.z; afr[ks][3]  = (__bf16)f00.w;
            afr[ks][4]  = (__bf16)f01.x; afr[ks][5]  = (__bf16)f01.y;
            afr[ks][6]  = (__bf16)f01.z; afr[ks][7]  = (__bf16)f01.w;
            afr[ks][8]  = (__bf16)f10.x; afr[ks][9]  = (__bf16)f10.y;
            afr[ks][10] = (__bf16)f10.z; afr[ks][11] = (__bf16)f10.w;
            afr[ks][12] = (__bf16)f11.x; afr[ks][13] = (__bf16)f11.y;
            afr[ks][14] = (__bf16)f11.z; afr[ks][15] = (__bf16)f11.w;
        }
        {
            float av = ali[((size_t)bt * S_ + s) * K_ + mrow];
            float dv = dst[((size_t)bt * S_ + s) * K_ + mrow];
            #pragma unroll
            for (int j = 0; j < 16; ++j) afr[4][j] = (__bf16)0.0f;
            afr[4][0] = (lane < 16) ? (__bf16)av : (__bf16)0.0f;  // feature 128
            afr[4][1] = (lane < 16) ? (__bf16)dv : (__bf16)0.0f;  // feature 129
        }

        // kn / vn projections: [16 x 160] * [160 x 128] via 8x5 WMMAs each
        #pragma unroll
        for (int proj = 0; proj < 2; ++proj) {
            const __bf16* wm = proj ? wv : wk;
            #pragma unroll
            for (int n = 0; n < 8; ++n) {
                v8f acc = {0.f, 0.f, 0.f, 0.f, 0.f, 0.f, 0.f, 0.f};
                #pragma unroll
                for (int ks = 0; ks < 5; ++ks) {
                    v16bf bfr = *(const v16bf*)(wm + (size_t)(n * 16 + ncol) * CP_ + ks * 32 + kseg);
                    acc = __builtin_amdgcn_wmma_f32_16x16x32_bf16(
                            false, afr[ks], false, bfr, (short)0, acc, false, false);
                }
                #pragma unroll
                for (int j = 0; j < 8; ++j) {
                    __bf16 v = (__bf16)acc[j];
                    if (proj) vnLds[w][base + j][n * 16 + ncol] = v;
                    else      knLds[w][base + j][n * 16 + ncol] = v;
                }
            }
        }
        __syncthreads();

        // Scores: each lane covers (h0, mrow) and (h0+2, mrow)
        const float lg = logf(swgt[((size_t)bt * S_ + s) * K_ + mrow] + 1e-6f);
        const int h0 = lane >> 4;              // 0 or 1
        float sc0 = 0.f, sc1 = 0.f;
        #pragma unroll
        for (int d = 0; d < D_; ++d) {
            sc0 += qtile[srow][h0 * D_ + d]       * (float)knLds[w][mrow][h0 * D_ + d];
            sc1 += qtile[srow][(h0 + 2) * D_ + d] * (float)knLds[w][mrow][(h0 + 2) * D_ + d];
        }
        sc0 = sc0 * rscale + lg;
        sc1 = sc1 * rscale + lg;
        // softmax over the 16 neighbors (16-lane groups; wave32 shuffles)
        float m0 = sc0, m1 = sc1;
        #pragma unroll
        for (int off = 8; off >= 1; off >>= 1) {
            m0 = fmaxf(m0, __shfl_xor(m0, off, 32));
            m1 = fmaxf(m1, __shfl_xor(m1, off, 32));
        }
        float e0 = __expf(sc0 - m0), e1 = __expf(sc1 - m1);
        float z0 = e0, z1 = e1;
        #pragma unroll
        for (int off = 8; off >= 1; off >>= 1) {
            z0 += __shfl_xor(z0, off, 32);
            z1 += __shfl_xor(z1, off, 32);
        }
        attnLds[w][h0][mrow]     = e0 / z0;
        attnLds[w][h0 + 2][mrow] = e1 / z1;
        __syncthreads();

        // out_s[c] = sum_k attn[h(c)][k] * vn[k][c]; lane owns c = lane+32j
        #pragma unroll
        for (int j = 0; j < 4; ++j) {
            const int c = lane + 32 * j;       // h = j
            float acc = 0.f;
            #pragma unroll
            for (int k = 0; k < K_; ++k)
                acc += attnLds[w][j][k] * (float)vnLds[w][k][c];
            otile[srow][c] = acc;
        }
        __syncthreads();
    }

    // ---- Stage 5: Y = otile * Wp^T + bp via WMMA, scatter to [B,S,T,C] ----
    {
        v16bf aof[4];
        #pragma unroll
        for (int ks = 0; ks < 4; ++ks) {
            #pragma unroll
            for (int j = 0; j < 8; ++j) {
                aof[ks][j]     = (__bf16)otile[mrow][ks * 32 + kseg + j];
                aof[ks][8 + j] = (__bf16)otile[mrow][ks * 32 + kseg + 8 + j];
            }
        }
        #pragma unroll
        for (int nt = 0; nt < 2; ++nt) {
            const int n = w * 2 + nt;
            v8f acc = {0.f, 0.f, 0.f, 0.f, 0.f, 0.f, 0.f, 0.f};
            #pragma unroll
            for (int ks = 0; ks < 4; ++ks) {
                v16bf bfr = *(const v16bf*)(wp + (size_t)(n * 16 + ncol) * C_ + ks * 32 + kseg);
                acc = __builtin_amdgcn_wmma_f32_16x16x32_bf16(
                        false, aof[ks], false, bfr, (short)0, acc, false, false);
            }
            const float bv = bp[n * 16 + ncol];
            #pragma unroll
            for (int j = 0; j < 8; ++j) {
                const int sr = base + j;       // s row within tile
                out[((size_t)(b * S_ + s0 + sr) * T_ + t) * C_ + n * 16 + ncol] = acc[j] + bv;
            }
        }
    }
}

// ---------------------------------------------------------------------------
extern "C" void kernel_launch(void* const* d_in, const int* in_sizes, int n_in,
                              void* d_out, int out_size, void* d_ws, size_t ws_size,
                              hipStream_t stream) {
    const float* x    = (const float*)d_in[0];
    const int*   sidx = (const int*)  d_in[1];   // spatial_idx (int32 on device)
    const float* swgt = (const float*)d_in[2];
    const float* ali  = (const float*)d_in[3];
    const float* dst  = (const float*)d_in[4];
    const float* Wq   = (const float*)d_in[5];
    const float* Wk   = (const float*)d_in[6];
    const float* Wv   = (const float*)d_in[7];
    const float* Wp   = (const float*)d_in[8];
    const float* bp   = (const float*)d_in[9];
    float*  out = (float*)d_out;
    __bf16* wbf = (__bf16*)d_ws;                 // 147456 bytes of scratch

    prep_weights_kernel<<<(WTOT + 255) / 256, 256, 0, stream>>>(Wq, Wk, Wv, Wp, wbf);
    spatial_attn_kernel<<<BT_ * NSTILE, 128, 0, stream>>>(
        x, sidx, swgt, ali, dst, wbf, bp, out);
}
